// ViLLayer_86492051407243
// MI455X (gfx1250) — compile-verified, matrix operand order checked by reading, not run on hardware
//
#include <hip/hip_runtime.h>
#include <cstdint>
#include <cstddef>

#define NB 8
#define SEQ 1024
#define DIMX 192
#define INNER 288
#define NHM 4
#define DHM 72

typedef __attribute__((ext_vector_type(16))) __bf16 v16bf;
typedef __attribute__((ext_vector_type(8)))  float  v8f;
typedef __attribute__((ext_vector_type(4)))  unsigned int u32x4;

union Frag16 { v16bf v; u32x4 q[2]; };

__device__ __forceinline__ unsigned short f2bf(float f) {
  unsigned int u = __float_as_uint(f);
  u += 0x7FFFu + ((u >> 16) & 1u);          // round-to-nearest-even
  return (unsigned short)(u >> 16);
}

__device__ __forceinline__ v8f wmma_bf16(const Frag16& a, const Frag16& b, v8f c) {
  return __builtin_amdgcn_wmma_f32_16x16x32_bf16(false, a.v, false, b.v, (short)0, c, false, false);
}

__device__ __forceinline__ float hsum16(float x) {   // butterfly within 16-lane half
  x += __shfl_xor(x, 1, 32);
  x += __shfl_xor(x, 2, 32);
  x += __shfl_xor(x, 4, 32);
  x += __shfl_xor(x, 8, 32);
  return x;
}

// ---------------- f32 -> bf16 convert ----------------
__global__ void cvt_bf16(const float* __restrict__ in, unsigned short* __restrict__ out, int n) {
  int i = blockIdx.x * 256 + threadIdx.x;
  if (i < n) out[i] = f2bf(in[i]);
}

// ---------------- bf16 GEMM: out[M,N] = A[M,K] * W[N,K]^T ----------------
// block = 128 threads = 4 waves computing a 64x64 tile; wave w owns rows m0+16w.
// The 64x32 W slab per k-step is staged into LDS with async copies (double-buffered);
// A fragments are direct contiguous global_load_b128 (reused by 4 WMMAs each).
// Requires M%64==0, N%64==0, K%32==0, grid exact.
__global__ void gemm_bf16_k(const unsigned short* __restrict__ A,
                            const unsigned short* __restrict__ W,
                            float* __restrict__ out, int M, int N, int K) {
  __shared__ __align__(16) unsigned short wtile[2][64 * 32];
  const int tid = threadIdx.x;
  const int wave = tid >> 5, lane = tid & 31;
  const int half = lane >> 4, l16 = lane & 15;
  const int ngroups = N >> 6;
  const int m0 = (blockIdx.x / ngroups) << 6;
  const int n0 = (blockIdx.x % ngroups) << 6;
  const int ksteps = K >> 5;
  const unsigned ldsbase = (unsigned)(uintptr_t)(&wtile[0][0]);  // low 32b = LDS offset

  // each thread copies two 16B chunks per stage: chunk j -> W row n0+j/4, k-seg 8*(j%4)
  const int j0 = tid, j1 = tid + 128;
  const unsigned short* g0base = W + (size_t)(n0 + (j0 >> 2)) * K + 8 * (j0 & 3);
  const unsigned short* g1base = W + (size_t)(n0 + (j1 >> 2)) * K + 8 * (j1 & 3);

#define ISSUE_STAGE(ks, buf)                                                           \
  {                                                                                    \
    unsigned l0 = ldsbase + (unsigned)(buf) * 4096u + (unsigned)j0 * 16u;              \
    unsigned l1 = ldsbase + (unsigned)(buf) * 4096u + (unsigned)j1 * 16u;              \
    unsigned long long a0 = (unsigned long long)(uintptr_t)(g0base + ((ks) << 5));     \
    unsigned long long a1 = (unsigned long long)(uintptr_t)(g1base + ((ks) << 5));     \
    asm volatile("global_load_async_to_lds_b128 %0, %1, off" ::"v"(l0), "v"(a0)        \
                 : "memory");                                                          \
    asm volatile("global_load_async_to_lds_b128 %0, %1, off" ::"v"(l1), "v"(a1)        \
                 : "memory");                                                          \
  }

  ISSUE_STAGE(0, 0)

  const v8f z8 = {0.f,0.f,0.f,0.f,0.f,0.f,0.f,0.f};
  v8f acc[4];
#pragma unroll
  for (int i = 0; i < 4; ++i) acc[i] = z8;
  const unsigned short* arow = A + (size_t)(m0 + wave * 16 + l16) * K;

  for (int ks = 0; ks < ksteps; ++ks) {
    const int buf = ks & 1;
    if (ks + 1 < ksteps) {
      ISSUE_STAGE(ks + 1, buf ^ 1)
      asm volatile("s_wait_asynccnt 0x2" ::: "memory");   // stage ks landed, ks+1 in flight
    } else {
      asm volatile("s_wait_asynccnt 0x0" ::: "memory");
    }
    __syncthreads();
    const int k0 = (ks << 5) + 8 * half;
    Frag16 a;
    a.q[0] = *(const u32x4*)(arow + k0);
    a.q[1] = *(const u32x4*)(arow + k0 + 16);
#pragma unroll
    for (int st = 0; st < 4; ++st) {
      const unsigned short* wl = &wtile[buf][(st * 16 + l16) * 32 + 8 * half];
      Frag16 b;
      b.q[0] = *(const u32x4*)(wl);
      b.q[1] = *(const u32x4*)(wl + 16);
      acc[st] = wmma_bf16(a, b, acc[st]);
    }
    __syncthreads();   // protect buf before stage ks+2 overwrites it
  }
#undef ISSUE_STAGE

#pragma unroll
  for (int st = 0; st < 4; ++st) {
    const int col = n0 + st * 16 + l16;
#pragma unroll
    for (int v = 0; v < 8; ++v)
      out[(size_t)(m0 + wave * 16 + 8 * half + v) * N + col] = acc[st][v];
  }
}

// ---------------- fused depthwise conv3x3 + SiLU + per-head qkv + gate dots ----------------
__global__ void conv_qkv_gates(const float* __restrict__ y, const float* __restrict__ convw,
                               const float* __restrict__ qw, const float* __restrict__ kw,
                               const float* __restrict__ vw, const float* __restrict__ igw,
                               const float* __restrict__ igb, const float* __restrict__ fgw,
                               const float* __restrict__ fgb,
                               float* __restrict__ xa_out,
                               unsigned short* __restrict__ qbf, unsigned short* __restrict__ kbf,
                               unsigned short* __restrict__ vtb,
                               float* __restrict__ ig, float* __restrict__ fg) {
  __shared__ float xaL[INNER], xmL[INNER], qL[INNER], kL[INNER], vL[INNER];
  __shared__ float red[8 * 128];
  const int bs = blockIdx.x;
  const int b = bs >> 10, s = bs & 1023;
  const int hh = s >> 5, ww = s & 31;
  const int tid = threadIdx.x;

  for (int c = tid; c < INNER; c += 128) {
    float a = 0.f;
#pragma unroll
    for (int dy = 0; dy < 3; ++dy) {
      const int hy = hh + dy - 1;
      if ((unsigned)hy < 32u) {
#pragma unroll
        for (int dx = 0; dx < 3; ++dx) {
          const int wx = ww + dx - 1;
          if ((unsigned)wx < 32u)
            a += convw[(dy * 3 + dx) * INNER + c] *
                 y[((size_t)(b * SEQ) + hy * 32 + wx) * (2 * INNER) + c];
        }
      }
    }
    const float xm = y[((size_t)(b * SEQ) + s) * (2 * INNER) + c];
    const float sa = a / (1.f + __expf(-a));      // silu(conv)
    xaL[c] = sa; xmL[c] = xm;
    xa_out[((size_t)(b * SEQ) + s) * INNER + c] = sa;
  }
  __syncthreads();

  for (int c = tid; c < INNER; c += 128) {
    const int hq = c >> 2, o = c & 3;
    float q = 0.f, k = 0.f, v = 0.f;
#pragma unroll
    for (int d = 0; d < 4; ++d) {
      q += xaL[(hq << 2) + d] * qw[hq * 16 + o * 4 + d];
      k += xaL[(hq << 2) + d] * kw[hq * 16 + o * 4 + d];
      v += xmL[(hq << 2) + d] * vw[hq * 16 + o * 4 + d];
    }
    qL[c] = q; kL[c] = k; vL[c] = v;
    const size_t base = ((size_t)(b * SEQ) + s) * INNER + c;
    qbf[base] = f2bf(q);
    kbf[base] = f2bf(k);
    vtb[((size_t)(b * INNER) + c) * SEQ + s] = f2bf(v);   // transposed V for contiguous B-frags
  }
  __syncthreads();

  float pacc[8] = {0.f,0.f,0.f,0.f,0.f,0.f,0.f,0.f};
  for (int c = tid; c < INNER; c += 128) {
    const float q = qL[c], k = kL[c], v = vL[c];
#pragma unroll
    for (int n = 0; n < 4; ++n) {
      pacc[n]     += q * igw[n * (3*INNER) + c] + k * igw[n * (3*INNER) + INNER + c] + v * igw[n * (3*INNER) + 2*INNER + c];
      pacc[4 + n] += q * fgw[n * (3*INNER) + c] + k * fgw[n * (3*INNER) + INNER + c] + v * fgw[n * (3*INNER) + 2*INNER + c];
    }
  }
#pragma unroll
  for (int j = 0; j < 8; ++j) red[j * 128 + tid] = pacc[j];
  __syncthreads();
  if (tid < 8) {
    float t = 0.f;
    for (int i = 0; i < 128; ++i) t += red[tid * 128 + i];
    const int n = tid & 3;
    if (tid < 4) ig[((size_t)(b * NHM) + n) * SEQ + s] = t + igb[n];
    else         fg[((size_t)(b * NHM) + n) * SEQ + s] = t + fgb[n];
  }
}

// ---------------- per-(b,head) scans: lfc = cumsum(logsigmoid(fg)); cmax = cummax(ig - lfc) ----------------
__global__ void scan_kernel(const float* __restrict__ fg, const float* __restrict__ ig,
                            float* __restrict__ rterm, float* __restrict__ cterm,
                            float* __restrict__ nexp) {
  __shared__ float buf[SEQ];
  const int bn = blockIdx.x;
  const int t = threadIdx.x;
  const float f = fg[(size_t)bn * SEQ + t];
  const float lf = fminf(f, 0.f) - log1pf(__expf(-fabsf(f)));  // log_sigmoid
  buf[t] = lf;
  __syncthreads();
  for (int off = 1; off < SEQ; off <<= 1) {
    float add = (t >= off) ? buf[t - off] : 0.f;
    __syncthreads();
    buf[t] += add;
    __syncthreads();
  }
  const float lfc = buf[t];
  const float c = ig[(size_t)bn * SEQ + t] - lfc;
  __syncthreads();
  buf[t] = c;
  __syncthreads();
  for (int off = 1; off < SEQ; off <<= 1) {
    float m = (t >= off) ? buf[t - off] : -3.0e38f;
    __syncthreads();
    buf[t] = fmaxf(buf[t], m);
    __syncthreads();
  }
  const float cmax = buf[t];
  rterm[(size_t)bn * SEQ + t] = -cmax;                 // lfc - maxD
  cterm[(size_t)bn * SEQ + t] = c;                     // ig - lfc
  nexp[(size_t)bn * SEQ + t]  = __expf(-(lfc + cmax)); // exp(-maxD)
}

// ---------------- flash-style causal mLSTM attention + LayerNorm ----------------
// one wave per 16-row tile; 4 waves / block, fully independent
__global__ void attn_kernel(const unsigned short* __restrict__ qbf,
                            const unsigned short* __restrict__ kbf,
                            const unsigned short* __restrict__ vt,
                            const float* __restrict__ rterm, const float* __restrict__ cterm,
                            const float* __restrict__ nexp, const float* __restrict__ norm_w,
                            float* __restrict__ hn) {
  __shared__ __align__(16) unsigned short ldsT[4][16 * 32];
  const int wave = threadIdx.x >> 5, lane = threadIdx.x & 31;
  const int id = blockIdx.x * 4 + wave;
  const int b = id >> 8;
  const int n = (id >> 6) & 3;
  const int r0 = (id & 63) << 4;
  const int half = lane >> 4, l16 = lane & 15;
  const u32x4 z4 = {0u, 0u, 0u, 0u};
  const v8f z8 = {0.f,0.f,0.f,0.f,0.f,0.f,0.f,0.f};

  // Q fragments: 16 rows x K=96 (zero-padded past 72)
  Frag16 qf[3];
  {
    const unsigned short* qrow = qbf + ((size_t)(b * SEQ + r0 + l16)) * INNER + n * DHM;
#pragma unroll
    for (int ks = 0; ks < 3; ++ks) {
      const int k0 = ks * 32 + 8 * half;
      qf[ks].q[0] = (k0 < DHM)      ? *(const u32x4*)(qrow + k0)      : z4;
      qf[ks].q[1] = (k0 + 16 < DHM) ? *(const u32x4*)(qrow + k0 + 16) : z4;
    }
  }
  const int bnS = (b * NHM + n) * SEQ;
  float rt8[8];
#pragma unroll
  for (int v = 0; v < 8; ++v) rt8[v] = rterm[bnS + r0 + 8 * half + v];

  v8f acc[5];
#pragma unroll
  for (int i = 0; i < 5; ++i) acc[i] = z8;
  float rs[8] = {0.f,0.f,0.f,0.f,0.f,0.f,0.f,0.f};

  unsigned short* tile = ldsT[wave];
  const float scale = 0.1178511301977579f;  // 1/sqrt(72)
  const int ntiles = (r0 >> 5) + 1;

  for (int ct = 0; ct < ntiles; ++ct) {
    const int c0 = ct << 5;
    const float ctA = cterm[bnS + c0 + l16];
    const float ctB = cterm[bnS + c0 + 16 + l16];
    v8f s0 = z8, s1 = z8;
    const unsigned short* krowA = kbf + ((size_t)(b * SEQ + c0 + l16)) * INNER + n * DHM;
    const unsigned short* krowB = krowA + (size_t)16 * INNER;
#pragma unroll
    for (int ks = 0; ks < 3; ++ks) {
      const int k0 = ks * 32 + 8 * half;
      Frag16 kf;
      kf.q[0] = (k0 < DHM)      ? *(const u32x4*)(krowA + k0)      : z4;
      kf.q[1] = (k0 + 16 < DHM) ? *(const u32x4*)(krowA + k0 + 16) : z4;
      s0 = wmma_bf16(qf[ks], kf, s0);
      kf.q[0] = (k0 < DHM)      ? *(const u32x4*)(krowB + k0)      : z4;
      kf.q[1] = (k0 + 16 < DHM) ? *(const u32x4*)(krowB + k0 + 16) : z4;
      s1 = wmma_bf16(qf[ks], kf, s1);
    }
    // decay weight, causal mask, unnormalized row sums; stage weighted tile to LDS as bf16
#pragma unroll
    for (int v = 0; v < 8; ++v) {
      const int srow = r0 + 8 * half + v;
      const int tA = c0 + l16, tB = c0 + 16 + l16;
      const float w0 = (tA <= srow) ? s0[v] * scale * __expf(rt8[v] + ctA) : 0.f;
      const float w1 = (tB <= srow) ? s1[v] * scale * __expf(rt8[v] + ctB) : 0.f;
      rs[v] += w0 + w1;
      tile[(8 * half + v) * 32 + l16]      = f2bf(w0);
      tile[(8 * half + v) * 32 + 16 + l16] = f2bf(w1);
    }
    // reload as WMMA A-fragment (same-wave LDS ops are in-order)
    Frag16 cf;
    cf.q[0] = *(const u32x4*)(tile + l16 * 32 + 8 * half);
    cf.q[1] = *(const u32x4*)(tile + l16 * 32 + 16 + 8 * half);
    // out(16x80) += C(16x32) x V(32x80), V pad cols are zero
#pragma unroll
    for (int nt = 0; nt < 5; ++nt) {
      const int dh = nt * 16 + l16;
      Frag16 vf;
      if (dh < DHM) {
        const unsigned short* vrow = vt + ((size_t)(b * INNER) + n * DHM + dh) * SEQ + c0;
        vf.q[0] = *(const u32x4*)(vrow + 8 * half);
        vf.q[1] = *(const u32x4*)(vrow + 16 + 8 * half);
      } else { vf.q[0] = z4; vf.q[1] = z4; }
      acc[nt] = wmma_bf16(cf, vf, acc[nt]);
    }
  }

  // normalizer: max(|row sum|, exp(-maxD)) + eps
  float inv[8];
#pragma unroll
  for (int v = 0; v < 8; ++v) {
    const float sum = hsum16(rs[v]);
    const float nx = nexp[bnS + r0 + 8 * half + v];
    inv[v] = 1.f / (fmaxf(fabsf(sum), nx) + 1e-6f);
  }
#pragma unroll
  for (int nt = 0; nt < 5; ++nt)
#pragma unroll
    for (int v = 0; v < 8; ++v) acc[nt][v] *= inv[v];

  // LayerNorm over dh<72 (pad cols are exact zeros)
  float mu[8], rstd[8];
  {
    float s1[8] = {0.f,0.f,0.f,0.f,0.f,0.f,0.f,0.f};
    float s2[8] = {0.f,0.f,0.f,0.f,0.f,0.f,0.f,0.f};
#pragma unroll
    for (int nt = 0; nt < 5; ++nt) {
      const bool dv = (nt * 16 + l16) < DHM;
#pragma unroll
      for (int v = 0; v < 8; ++v) {
        const float x = dv ? acc[nt][v] : 0.f;
        s1[v] += x; s2[v] += x * x;
      }
    }
#pragma unroll
    for (int v = 0; v < 8; ++v) {
      const float a = hsum16(s1[v]);
      const float q = hsum16(s2[v]);
      const float m = a * (1.f / 72.f);
      mu[v] = m;
      rstd[v] = rsqrtf(q * (1.f / 72.f) - m * m + 1e-5f);
    }
  }
#pragma unroll
  for (int nt = 0; nt < 5; ++nt) {
    const int dh = nt * 16 + l16;
    if (dh < DHM) {
      const float nw = norm_w[n * DHM + dh];
#pragma unroll
      for (int v = 0; v < 8; ++v) {
        const int srow = r0 + 8 * half + v;
        hn[((size_t)(b * SEQ + srow)) * INNER + n * DHM + dh] = (acc[nt][v] - mu[v]) * rstd[v] * nw;
      }
    }
  }
}

// ---------------- h_state = (hn + skip*xa) * silu(z), emit bf16 for proj_down ----------------
__global__ void hstate_kernel(const float* __restrict__ hn, const float* __restrict__ xa,
                              const float* __restrict__ skip, const float* __restrict__ y,
                              unsigned short* __restrict__ hsbf, int n) {
  const int i = blockIdx.x * 256 + threadIdx.x;
  if (i >= n) return;
  const int c = i % INNER;
  const int bs = i / INNER;
  const float z = y[(size_t)bs * (2 * INNER) + INNER + c];
  const float hs = (hn[i] + skip[c] * xa[i]) * (z / (1.f + __expf(-z)));
  hsbf[i] = f2bf(hs);
}

extern "C" void kernel_launch(void* const* d_in, const int* in_sizes, int n_in,
                              void* d_out, int out_size, void* d_ws, size_t ws_size,
                              hipStream_t stream) {
  (void)in_sizes; (void)n_in; (void)out_size; (void)ws_size;
  const float* x     = (const float*)d_in[0];
  const float* wup   = (const float*)d_in[1];
  const float* qw    = (const float*)d_in[2];
  const float* kw    = (const float*)d_in[3];
  const float* vw    = (const float*)d_in[4];
  const float* convw = (const float*)d_in[5];
  const float* igw   = (const float*)d_in[6];
  const float* igb   = (const float*)d_in[7];
  const float* fgw   = (const float*)d_in[8];
  const float* fgb   = (const float*)d_in[9];
  const float* normw = (const float*)d_in[10];
  const float* skip  = (const float*)d_in[11];
  const float* wdown = (const float*)d_in[12];
  float* outp = (float*)d_out;

  char* p = (char*)d_ws;
  auto take = [&](size_t bytes) -> void* {
    char* r = p;
    p += (bytes + 255) & ~(size_t)255;
    return (void*)r;
  };
  unsigned short* xbf   = (unsigned short*)take((size_t)NB * SEQ * DIMX * 2);
  unsigned short* wupbf = (unsigned short*)take((size_t)2 * INNER * DIMX * 2);
  unsigned short* wdnbf = (unsigned short*)take((size_t)DIMX * INNER * 2);
  float* y   = (float*)take((size_t)NB * SEQ * 2 * INNER * 4);
  float* xa  = (float*)take((size_t)NB * SEQ * INNER * 4);
  unsigned short* qbf = (unsigned short*)take((size_t)NB * SEQ * INNER * 2);
  unsigned short* kbf = (unsigned short*)take((size_t)NB * SEQ * INNER * 2);
  unsigned short* vtb = (unsigned short*)take((size_t)NB * INNER * SEQ * 2);
  float* ig    = (float*)take((size_t)NB * NHM * SEQ * 4);
  float* fg    = (float*)take((size_t)NB * NHM * SEQ * 4);
  float* rterm = (float*)take((size_t)NB * NHM * SEQ * 4);
  float* cterm = (float*)take((size_t)NB * NHM * SEQ * 4);
  float* nexp  = (float*)take((size_t)NB * NHM * SEQ * 4);
  float* hn    = (float*)take((size_t)NB * SEQ * INNER * 4);
  unsigned short* hsbf = (unsigned short*)take((size_t)NB * SEQ * INNER * 2);

  { int n = NB * SEQ * DIMX;   cvt_bf16<<<(n + 255) / 256, 256, 0, stream>>>(x, xbf, n); }
  { int n = 2 * INNER * DIMX;  cvt_bf16<<<(n + 255) / 256, 256, 0, stream>>>(wup, wupbf, n); }
  { int n = DIMX * INNER;      cvt_bf16<<<(n + 255) / 256, 256, 0, stream>>>(wdown, wdnbf, n); }

  { // proj_up: (8192x192) x (576x192)^T -> y (8192x576); 64x64 tiles
    int blocks = (NB * SEQ / 64) * (2 * INNER / 64);
    gemm_bf16_k<<<blocks, 128, 0, stream>>>(xbf, wupbf, y, NB * SEQ, 2 * INNER, DIMX);
  }

  conv_qkv_gates<<<NB * SEQ, 128, 0, stream>>>(y, convw, qw, kw, vw, igw, igb, fgw, fgb,
                                               xa, qbf, kbf, vtb, ig, fg);

  scan_kernel<<<NB * NHM, SEQ, 0, stream>>>(fg, ig, rterm, cterm, nexp);

  attn_kernel<<<NB * NHM * (SEQ / 16) / 4, 128, 0, stream>>>(qbf, kbf, vtb, rterm, cterm,
                                                             nexp, normw, hn);

  { int n = NB * SEQ * INNER;
    hstate_kernel<<<(n + 255) / 256, 256, 0, stream>>>(hn, xa, skip, y, hsbf, n); }

  { // proj_down: (8192x288) x (192x288)^T -> out (8192x192); 64x64 tiles
    int blocks = (NB * SEQ / 64) * (DIMX / 64);
    gemm_bf16_k<<<blocks, 128, 0, stream>>>(hsbf, wdnbf, outp, NB * SEQ, DIMX, INNER);
  }
}
